// MultiHeadSelfAttention_23081154248831
// MI455X (gfx1250) — compile-verified, matrix operand order checked by reading, not run on hardware
//
#include <hip/hip_runtime.h>

#define BATCH 4
#define SLEN  2048
#define EMB   1024
#define NH    16
#define HD    64

typedef __bf16 bf16;
typedef __attribute__((ext_vector_type(16))) __bf16 v16bf;
typedef __attribute__((ext_vector_type(8)))  __bf16 v8bf;
typedef __attribute__((ext_vector_type(8)))  float  v8f;
typedef __attribute__((ext_vector_type(4)))  unsigned int u32x4;
typedef __attribute__((ext_vector_type(8)))  int i32x8;
typedef __attribute__((ext_vector_type(4)))  int i32x4;

// ---------------------------------------------------------------------------
// Fragment loader: row-major bf16 tile -> 16x32 A-style WMMA operand.
// ISA layout (CDNA5 ISA, 16-bit A 16x32): lane L holds row m=L&15;
// lanes 0-15: K = {0..7, 16..23}; lanes 16-31: K = {8..15, 24..31}.
// Works for both global and LDS pointers (addrspace inferred).
// ---------------------------------------------------------------------------
__device__ __forceinline__ v16bf load_frag(const bf16* base, int ld) {
  const int lane = threadIdx.x & 31;
  const int m    = lane & 15;
  const int hi   = lane >> 4;
  const bf16* p  = base + (size_t)m * ld + hi * 8;
  v8bf lo = *(const v8bf*)(p);
  v8bf hh = *(const v8bf*)(p + 16);
  v16bf f;
#pragma unroll
  for (int i = 0; i < 8; ++i) { f[i] = lo[i]; f[8 + i] = hh[i]; }
  return f;
}

__device__ __forceinline__ v8f wmma_bf16(v16bf a, v16bf b, v8f c) {
  return __builtin_amdgcn_wmma_f32_16x16x32_bf16(
      /*neg_a=*/false, a, /*neg_b=*/false, b,
      /*c_mod=*/(short)0, c, /*reuse_a=*/false, /*reuse_b=*/false);
}

// ---------------------------------------------------------------------------
// TDM helpers (CDNA5 ISA/§8).
// 2-D tile load, data_size = 2 bytes. All operands must be wave-uniform.
//   group0: [1:0]=count=1 | lds_addr | global_addr[56:0] | type=2
//   group1: data_size=1(2B), tensor_dim0/1 (= tile dims, no OOB clipping),
//           tile_dim0/1, tensor_dim0_stride (elements)
// This toolchain's builtin takes 6 operands (probe-confirmed clang-23 form):
//   (u32x4 g0, i32x8 g1, i32x4 g2, i32x4 g3, i32x8 extra, i32 cpol)
// ---------------------------------------------------------------------------
typedef __attribute__((address_space(3))) char lds_char;

__device__ __forceinline__ unsigned lds_offset(void* p) {
  return (unsigned)(unsigned long long)(lds_char*)p;   // AS3 ptr == LDS byte offset
}

__device__ __forceinline__ void tdm_load_2d(unsigned lds_addr, const bf16* gptr,
                                            unsigned tile_d0, unsigned tile_d1,
                                            unsigned long long stride0_elems) {
  unsigned long long ga = (unsigned long long)gptr;
  u32x4 g0;
  g0[0] = 1u;                                            // count=1, user mode
  g0[1] = lds_addr;                                      // LDS byte address
  g0[2] = (unsigned)ga;                                  // global_addr[31:0]
  g0[3] = (unsigned)((ga >> 32) & 0x01ffffffu) | (2u << 30); // addr[56:32] | type=2

  i32x8 g1;
  g1[0] = (int)(1u << 16);                               // data_size = 2 bytes
  g1[1] = (int)((tile_d0 & 0xffffu) << 16);              // tensor_dim0[15:0]
  g1[2] = (int)(((tile_d0 >> 16) & 0xffffu) |            // tensor_dim0[31:16]
                ((tile_d1 & 0xffffu) << 16));            // tensor_dim1[15:0]
  g1[3] = (int)(((tile_d1 >> 16) & 0xffffu) |            // tensor_dim1[31:16]
                ((tile_d0 & 0xffffu) << 16));            // tile_dim0
  g1[4] = (int)(tile_d1 & 0xffffu);                      // tile_dim1 (tile_dim2=0)
  g1[5] = (int)(stride0_elems & 0xffffffffu);            // dim0_stride[31:0]
  g1[6] = (int)((stride0_elems >> 32) & 0xffffu);        // dim0_stride[47:32]
  g1[7] = 0;

  i32x4 g2; g2[0] = 0; g2[1] = 0; g2[2] = 0; g2[3] = 0;  // dims 2/3 unused
  i32x4 g3; g3[0] = 0; g3[1] = 0; g3[2] = 0; g3[3] = 0;  // dim 4 unused
  i32x8 gx;                                              // unused trailing group
#pragma unroll
  for (int i = 0; i < 8; ++i) gx[i] = 0;
  __builtin_amdgcn_tensor_load_to_lds(g0, g1, g2, g3, gx, 0);
}

// ---------------------------------------------------------------------------
// Kernel 1a: fp32 -> bf16 elementwise convert
// ---------------------------------------------------------------------------
__global__ void cvt_bf16_kernel(const float* __restrict__ in,
                                bf16* __restrict__ out, int n) {
  int i = blockIdx.x * blockDim.x + threadIdx.x;
  if (i < n) out[i] = (bf16)in[i];
}

// ---------------------------------------------------------------------------
// Kernel 1b: W (E x E fp32, row-major) -> W^T (E x E bf16, row-major)
// ---------------------------------------------------------------------------
__global__ void transpose_bf16_kernel(const float* __restrict__ W,
                                      bf16* __restrict__ WT) {
  int idx = blockIdx.x * blockDim.x + threadIdx.x;   // over E*E
  int n = idx >> 10;
  int k = idx & (EMB - 1);
  WT[idx] = (bf16)W[(size_t)k * EMB + n];
}

// ---------------------------------------------------------------------------
// QKV epilogue: scatter one 16x16 f32 tile into Q/K/V^T bf16 layouts.
// ---------------------------------------------------------------------------
__device__ __forceinline__ void store_qkv_tile(const v8f& c, int rowBase,
                                               int colBase, int mode,
                                               const float* __restrict__ bias,
                                               bf16* __restrict__ Qm,
                                               bf16* __restrict__ Km,
                                               bf16* __restrict__ VTm) {
  const int lane = threadIdx.x & 31;
  const int n  = lane & 15;
  const int hi = lane >> 4;
  const int col = colBase + n;
  const float bval = bias[col];
  const int h = col >> 6;
  const int d = col & 63;
  const float qscale = 1.4426950408889634f / 8.0f;   // log2(e)/sqrt(D)
#pragma unroll
  for (int r = 0; r < 8; ++r) {
    int row = rowBase + r + 8 * hi;                  // token in [0, B*S)
    int b_  = row >> 11;
    int s_  = row & (SLEN - 1);
    float y = c[r] + bval;
    size_t bh = (size_t)(b_ * NH + h);
    if (mode == 0)      Qm[(bh * SLEN + s_) * HD + d] = (bf16)(y * qscale);
    else if (mode == 1) Km[(bh * SLEN + s_) * HD + d] = (bf16)y;
    else                VTm[(bh * HD + d) * SLEN + s_] = (bf16)y;
  }
}

// ---------------------------------------------------------------------------
// Kernel 2: fused QKV projection, 2x2 register tiling (32x32 per wave).
// grid = (colTiles32/8, rowTiles32, 3), block = 256 (8 waves).
// ---------------------------------------------------------------------------
__global__ void __launch_bounds__(256)
qkv_proj_kernel(const bf16* __restrict__ xb,
                const bf16* __restrict__ WqT, const bf16* __restrict__ WkT,
                const bf16* __restrict__ WvT,
                const float* __restrict__ bq, const float* __restrict__ bk,
                const float* __restrict__ bv,
                bf16* __restrict__ Qm, bf16* __restrict__ Km,
                bf16* __restrict__ VTm) {
  const int wave    = threadIdx.x >> 5;
  const int rowBase = blockIdx.y * 32;
  const int colBase = (blockIdx.x * 8 + wave) * 32;
  const int mode    = blockIdx.z;

  const bf16* WT; const float* bias;
  if (mode == 0)      { WT = WqT; bias = bq; }
  else if (mode == 1) { WT = WkT; bias = bk; }
  else                { WT = WvT; bias = bv; }

  const bf16* a0p = xb + (size_t)rowBase * EMB;
  const bf16* a1p = a0p + (size_t)16 * EMB;
  const bf16* b0p = WT + (size_t)colBase * EMB;
  const bf16* b1p = b0p + (size_t)16 * EMB;

  v8f c00 = {}, c01 = {}, c10 = {}, c11 = {};
#pragma unroll 2
  for (int k0 = 0; k0 < EMB; k0 += 32) {
    v16bf a0 = load_frag(a0p + k0, EMB);
    v16bf a1 = load_frag(a1p + k0, EMB);
    v16bf b0 = load_frag(b0p + k0, EMB);
    v16bf b1 = load_frag(b1p + k0, EMB);
    c00 = wmma_bf16(a0, b0, c00);
    c01 = wmma_bf16(a0, b1, c01);
    c10 = wmma_bf16(a1, b0, c10);
    c11 = wmma_bf16(a1, b1, c11);
  }
  store_qkv_tile(c00, rowBase,      colBase,      mode, bias, Qm, Km, VTm);
  store_qkv_tile(c01, rowBase,      colBase + 16, mode, bias, Qm, Km, VTm);
  store_qkv_tile(c10, rowBase + 16, colBase,      mode, bias, Qm, Km, VTm);
  store_qkv_tile(c11, rowBase + 16, colBase + 16, mode, bias, Qm, Km, VTm);
}

// ---------------------------------------------------------------------------
// Kernel 3: causal flash attention, one wave per block (TDM descriptors must
// be wave-uniform SGPRs). K/V^T tiles are staged into LDS by the Tensor Data
// Mover (double-buffered, s_wait_tensorcnt pipelined); WMMA fragments then
// come from LDS (ds_load_b128). Scores computed as S^T = K_tile @ Q^T so the
// score C-layout coincides with the A-fragment layout of P. Softmax in exp2
// domain (scale folded into Q at projection time).
// ---------------------------------------------------------------------------
__global__ void __launch_bounds__(32)
flash_attn_kernel(const bf16* __restrict__ Qm, const bf16* __restrict__ Km,
                  const bf16* __restrict__ VTm, bf16* __restrict__ Ob) {
  __shared__ bf16 sK[2][32 * HD];   // 32 keys x 64 d, row-major (4 KB each)
  __shared__ bf16 sV[2][HD * 32];   // 64 d-rows x 32 keys (4 KB each)

  const int lane  = threadIdx.x & 31;
  const int bh    = blockIdx.y;                 // 0..B*H-1
  const int b_    = bh >> 4;
  const int h     = bh & (NH - 1);
  const int qBase = blockIdx.x * 16;

  const int hi = lane >> 4;
  const int qq = qBase + (lane & 15);

  const bf16* Qbase = Qm + ((size_t)bh * SLEN + qBase) * HD;
  v16bf qf0 = load_frag(Qbase,      HD);        // d = 0..31
  v16bf qf1 = load_frag(Qbase + 32, HD);        // d = 32..63

  const bf16* Kb0 = Km  + (size_t)bh * SLEN * HD;
  const bf16* Vb0 = VTm + (size_t)bh * HD * SLEN;

  unsigned offK0 = lds_offset(&sK[0][0]);
  unsigned offK1 = lds_offset(&sK[1][0]);
  unsigned offV0 = lds_offset(&sV[0][0]);
  unsigned offV1 = lds_offset(&sV[1][0]);

  const int ktMax = (qBase + 15) >> 5;          // inclusive, causal bound

  // prologue: stage tile 0 (K: 32 rows x 64 contiguous; V^T: 64 rows x 32,
  // row stride = SLEN elements -- exactly TDM tensor_dim0_stride)
  tdm_load_2d(offK0, Kb0, HD, 32, HD);
  tdm_load_2d(offV0, Vb0, 32, HD, SLEN);

  float m_run = -1e30f, l_run = 0.0f;
  v8f o0 = {}, o1 = {}, o2 = {}, o3 = {};

  for (int kt = 0; kt <= ktMax; ++kt) {
    const int buf = kt & 1;
    if (kt < ktMax) {
      const int k1 = (kt + 1) * 32;
      // the target buffer was last *read* at iteration kt-1: drain DS reads
      asm volatile("s_wait_dscnt 0x0" ::: "memory");
      tdm_load_2d(buf ? offK0 : offK1, Kb0 + (size_t)k1 * HD, HD, 32, HD);
      tdm_load_2d(buf ? offV0 : offV1, Vb0 + k1, 32, HD, SLEN);
      // tile kt's two TDM ops are older than the two just issued
      asm volatile("s_wait_tensorcnt 0x2" ::: "memory");
    } else {
      asm volatile("s_wait_tensorcnt 0x0" ::: "memory");
    }
    const int k0 = kt * 32;
    const bf16* Kt = &sK[buf][0];
    const bf16* Vt = &sV[buf][0];

    // scores^T: C holds (key = r + 8*hi, query = lane&15)
    v16bf ka0 = load_frag(Kt,               HD);
    v16bf ka1 = load_frag(Kt + 32,          HD);
    v16bf kb0 = load_frag(Kt + 16 * HD,      HD);
    v16bf kb1 = load_frag(Kt + 16 * HD + 32, HD);
    v8f c0 = {}, c1 = {};
    c0 = wmma_bf16(ka0, qf0, c0);
    c0 = wmma_bf16(ka1, qf1, c0);
    c1 = wmma_bf16(kb0, qf0, c1);
    c1 = wmma_bf16(kb1, qf1, c1);

    // causal mask + per-query max over this key tile
    float s0[8], s1[8];
    float tmax = -1e30f;
#pragma unroll
    for (int r = 0; r < 8; ++r) {
      int kk = k0 + r + 8 * hi;
      s0[r] = (kk      <= qq) ? c0[r] : -1e30f;
      s1[r] = (kk + 16 <= qq) ? c1[r] : -1e30f;
      tmax = fmaxf(tmax, fmaxf(s0[r], s1[r]));
    }
    tmax = fmaxf(tmax, __shfl_xor(tmax, 16, 32));
    float m_new = fmaxf(m_run, tmax);
    float sf    = exp2f(m_run - m_new);

    float psum = 0.0f;
    v16bf pf;
#pragma unroll
    for (int r = 0; r < 8; ++r) {
      float p0 = exp2f(s0[r] - m_new);
      float p1 = exp2f(s1[r] - m_new);
      psum += p0 + p1;
      pf[r]     = (bf16)p0;
      pf[8 + r] = (bf16)p1;
    }
    psum += __shfl_xor(psum, 16, 32);
    l_run = l_run * sf + psum;
    m_run = m_new;

    // rescale O accumulators; O rows are q = r + 8*hi, stats live at lane q
#pragma unroll
    for (int r = 0; r < 8; ++r) {
      float fr = __shfl(sf, 8 * hi + r, 32);
      o0[r] *= fr; o1[r] *= fr; o2[r] *= fr; o3[r] *= fr;
    }

    // O += P @ V (B operand = rows of V^T tile, contiguous in LDS)
    v16bf vf0 = load_frag(Vt + 0 * 16 * 32, 32);
    v16bf vf1 = load_frag(Vt + 1 * 16 * 32, 32);
    v16bf vf2 = load_frag(Vt + 2 * 16 * 32, 32);
    v16bf vf3 = load_frag(Vt + 3 * 16 * 32, 32);
    o0 = wmma_bf16(pf, vf0, o0);
    o1 = wmma_bf16(pf, vf1, o1);
    o2 = wmma_bf16(pf, vf2, o2);
    o3 = wmma_bf16(pf, vf3, o3);
  }

  // normalize and store heads back into (B,S,E)
  float inv = 1.0f / l_run;
  const int n = lane & 15;
#pragma unroll
  for (int r = 0; r < 8; ++r) {
    float fr = __shfl(inv, 8 * hi + r, 32);
    int row = qBase + r + 8 * hi;
    size_t base = ((size_t)b_ * SLEN + row) * EMB + h * HD;
    Ob[base + 0  + n] = (bf16)(o0[r] * fr);
    Ob[base + 16 + n] = (bf16)(o1[r] * fr);
    Ob[base + 32 + n] = (bf16)(o2[r] * fr);
    Ob[base + 48 + n] = (bf16)(o3[r] * fr);
  }
}

// ---------------------------------------------------------------------------
// Kernel 4: output projection  out = Ob @ Wo + bo, 2x2 tiling, fp32 store.
// ---------------------------------------------------------------------------
__global__ void __launch_bounds__(256)
out_proj_kernel(const bf16* __restrict__ Ob, const bf16* __restrict__ WoT,
                const float* __restrict__ bo, float* __restrict__ out) {
  const int wave    = threadIdx.x >> 5;
  const int lane    = threadIdx.x & 31;
  const int rowBase = blockIdx.y * 32;
  const int colBase = (blockIdx.x * 8 + wave) * 32;

  const bf16* a0p = Ob + (size_t)rowBase * EMB;
  const bf16* a1p = a0p + (size_t)16 * EMB;
  const bf16* b0p = WoT + (size_t)colBase * EMB;
  const bf16* b1p = b0p + (size_t)16 * EMB;

  v8f c00 = {}, c01 = {}, c10 = {}, c11 = {};
#pragma unroll 2
  for (int k0 = 0; k0 < EMB; k0 += 32) {
    v16bf a0 = load_frag(a0p + k0, EMB);
    v16bf a1 = load_frag(a1p + k0, EMB);
    v16bf b0 = load_frag(b0p + k0, EMB);
    v16bf b1 = load_frag(b1p + k0, EMB);
    c00 = wmma_bf16(a0, b0, c00);
    c01 = wmma_bf16(a0, b1, c01);
    c10 = wmma_bf16(a1, b0, c10);
    c11 = wmma_bf16(a1, b1, c11);
  }

  const int n  = lane & 15;
  const int hi = lane >> 4;
  const float bv0 = bo[colBase + n];
  const float bv1 = bo[colBase + 16 + n];
#pragma unroll
  for (int r = 0; r < 8; ++r) {
    size_t r0 = (size_t)(rowBase + r + 8 * hi);
    size_t r1 = r0 + 16;
    out[r0 * EMB + colBase + n]      = c00[r] + bv0;
    out[r0 * EMB + colBase + 16 + n] = c01[r] + bv1;
    out[r1 * EMB + colBase + n]      = c10[r] + bv0;
    out[r1 * EMB + colBase + 16 + n] = c11[r] + bv1;
  }
}

// ---------------------------------------------------------------------------
// Host launcher
// ---------------------------------------------------------------------------
extern "C" void kernel_launch(void* const* d_in, const int* in_sizes, int n_in,
                              void* d_out, int out_size, void* d_ws,
                              size_t ws_size, hipStream_t stream) {
  const float* x  = (const float*)d_in[0];
  const float* Wq = (const float*)d_in[1];
  const float* bq = (const float*)d_in[2];
  const float* Wk = (const float*)d_in[3];
  const float* bk = (const float*)d_in[4];
  const float* Wv = (const float*)d_in[5];
  const float* bv = (const float*)d_in[6];
  const float* Wo = (const float*)d_in[7];
  const float* bo = (const float*)d_in[8];
  float* out = (float*)d_out;

  const size_t NTOK   = (size_t)BATCH * SLEN;             // 8192
  const size_t XB_B   = NTOK * EMB * sizeof(bf16);        // 16 MiB
  const size_t WT_B   = (size_t)EMB * EMB * sizeof(bf16); // 2 MiB
  const size_t HEAD_B = (size_t)BATCH * NH * SLEN * HD * sizeof(bf16);

  char* ws = (char*)d_ws;
  bf16* xb  = (bf16*)ws;  ws += XB_B;
  bf16* WqT = (bf16*)ws;  ws += WT_B;
  bf16* WkT = (bf16*)ws;  ws += WT_B;
  bf16* WvT = (bf16*)ws;  ws += WT_B;
  bf16* WoT = (bf16*)ws;  ws += WT_B;
  bf16* Qm  = (bf16*)ws;  ws += HEAD_B;
  bf16* Km  = (bf16*)ws;  ws += HEAD_B;
  bf16* VTm = (bf16*)ws;  ws += HEAD_B;
  bf16* Ob  = (bf16*)ws;  ws += XB_B;

  // 1) converts
  {
    int n = (int)(NTOK * EMB);
    cvt_bf16_kernel<<<n / 256, 256, 0, stream>>>(x, xb, n);
    int nw = EMB * EMB / 256;
    transpose_bf16_kernel<<<nw, 256, 0, stream>>>(Wq, WqT);
    transpose_bf16_kernel<<<nw, 256, 0, stream>>>(Wk, WkT);
    transpose_bf16_kernel<<<nw, 256, 0, stream>>>(Wv, WvT);
    transpose_bf16_kernel<<<nw, 256, 0, stream>>>(Wo, WoT);
  }

  // 2) fused QKV projection: 256 row tiles (32) x 32 col tiles (32) x {Q,K,V}
  qkv_proj_kernel<<<dim3(4, 256, 3), 256, 0, stream>>>(
      xb, WqT, WkT, WvT, bq, bk, bv, Qm, Km, VTm);

  // 3) flash attention: 128 query tiles x 64 (b,h) pairs, 1 wave/block (TDM)
  flash_attn_kernel<<<dim3(128, 64), 32, 0, stream>>>(Qm, Km, VTm, Ob);

  // 4) output projection
  out_proj_kernel<<<dim3(4, 256), 256, 0, stream>>>(Ob, WoT, bo, out);
}
